// GATLayer_39453569581493
// MI455X (gfx1250) — matrix operand order repaired and checked by reading, not compiled, not run on hardware
//
#include <hip/hip_runtime.h>
#include <hip/hip_bf16.h>

typedef _Float16 f16;
typedef __attribute__((ext_vector_type(2)))  float v2f;
typedef __attribute__((ext_vector_type(8)))  float v8f;
typedef __attribute__((ext_vector_type(8)))  f16   v8h;
typedef __attribute__((ext_vector_type(16))) f16   v16h;

#define N_NODES 8192
#define F_IN    512
#define F_OUT   256
#define ALPHA   0.2f
#define NEG_INF (-9.0e15f)
#define BK      128              // attention tile columns staged in LDS
#define PROW    136              // sm_p16 row stride (f16), 272B: 16B-aligned, conflict-free

// ---------------------------------------------------------------------------
// Kernel 1: h = x @ W in f32 WMMA 16x16x4 (full precision), plus an
// LDS-staged transpose producing h16T (256 x 8192, f16) so the attention
// matmul's B-fragments are contiguous 16B loads.
// Block = 256 thr (8 waves), one 16-row band; wave owns 16x32 of output.
// ---------------------------------------------------------------------------
__global__ __launch_bounds__(256) void gat_xw(const float* __restrict__ x,
                                              const float* __restrict__ W,
                                              float* __restrict__ h,
                                              f16*   __restrict__ h16T) {
  __shared__ __align__(16) f16 sm_h[16][F_OUT + 8];

  const int band = blockIdx.x;           // 0..511
  const int wave = threadIdx.x >> 5;     // 0..7
  const int lane = threadIdx.x & 31;
  const int l16  = lane & 15;
  const bool hi  = lane >= 16;
  const int r0   = band * 16;
  const int n0   = wave * 32;

  v8f acc0 = {};
  v8f acc1 = {};

  const float* xrow = x + (size_t)(r0 + l16) * F_IN + (hi ? 2 : 0);
  for (int k = 0; k < F_IN; k += 4) {
    v2f a;
    a.x = xrow[k];
    a.y = xrow[k + 1];
    const float* wb = W + (size_t)(k + (hi ? 2 : 0)) * F_OUT + l16;
    v2f b0, b1;
    b0.x = wb[n0];      b0.y = wb[F_OUT + n0];
    b1.x = wb[n0 + 16]; b1.y = wb[F_OUT + n0 + 16];
    acc0 = __builtin_amdgcn_wmma_f32_16x16x4_f32(false, a, false, b0,
                                                 (short)0, acc0, false, false);
    acc1 = __builtin_amdgcn_wmma_f32_16x16x4_f32(false, a, false, b1,
                                                 (short)0, acc1, false, false);
  }

  // D layout: VGPR r -> row r / r+8 (half-waves), col = lane&15
  const int lr  = hi ? 8 : 0;
  const int col = n0 + l16;
#pragma unroll
  for (int r = 0; r < 8; ++r) {
    h[(size_t)(r0 + lr + r) * F_OUT + col]      = acc0[r];
    h[(size_t)(r0 + lr + r) * F_OUT + col + 16] = acc1[r];
    sm_h[lr + r][col]      = (f16)acc0[r];
    sm_h[lr + r][col + 16] = (f16)acc1[r];
  }
  __syncthreads();

  // transpose out: thread t owns column n=t; 16 f16 contiguous (32B) per column
  {
    const int n = threadIdx.x;           // 0..255
    f16 tmp[16];
#pragma unroll
    for (int r = 0; r < 16; ++r) tmp[r] = sm_h[r][n];
    v8h* dst = (v8h*)(h16T + (size_t)n * N_NODES + r0);
    dst[0] = *(const v8h*)&tmp[0];
    dst[1] = *(const v8h*)&tmp[8];
  }
}

// ---------------------------------------------------------------------------
// Kernel 2: e_src[i] = h[i,:].a[0:256], e_dst[i] = h[i,:].a[256:512]
// ---------------------------------------------------------------------------
__global__ __launch_bounds__(256) void gat_ev(const float* __restrict__ h,
                                              const float* __restrict__ a,
                                              float* __restrict__ e_src,
                                              float* __restrict__ e_dst) {
  const int row  = blockIdx.x * 8 + (threadIdx.x >> 5);
  const int lane = threadIdx.x & 31;
  const float* hr = h + (size_t)row * F_OUT;

  float s0 = 0.f, s1 = 0.f;
#pragma unroll
  for (int q = 0; q < F_OUT / 32; ++q) {
    const int c = lane + 32 * q;
    const float v = hr[c];
    s0 += v * a[c];
    s1 += v * a[F_OUT + c];
  }
#pragma unroll
  for (int off = 16; off > 0; off >>= 1) {
    s0 += __shfl_xor(s0, off, 32);
    s1 += __shfl_xor(s1, off, 32);
  }
  if (lane == 0) { e_src[row] = s0; e_dst[row] = s1; }
}

// ---------------------------------------------------------------------------
// Kernel 3: fused masked softmax + (attention @ h) in f16 WMMA 16x16x32,
// f32 accumulation. Two passes over adj; probability tiles live only in LDS.
// Block = 256 thr (8 waves) -> 16-row output band; wave owns 32 output cols.
// ---------------------------------------------------------------------------
__global__ __launch_bounds__(256) void gat_attn(const f16*   __restrict__ h16T,
                                                const int*   __restrict__ adj,
                                                const float* __restrict__ e_src,
                                                const float* __restrict__ e_dst,
                                                float* __restrict__ out) {
  __shared__ float sm_m[16];
  __shared__ float sm_s[16];
  __shared__ __align__(16) f16 sm_p16[16][PROW];

  const int band = blockIdx.x;     // 0..511
  const int r0   = band * 16;
  const int tid  = threadIdx.x;
  const int rr   = tid >> 4;       // row within band
  const int tt   = tid & 15;       // 16 threads per row
  const int row  = r0 + rr;
  const float es = e_src[row];
  const size_t adjrow = (size_t)row * N_NODES;

  // ---------------- Pass 1: online softmax stats (vectorized) ----------------
  float m = -3.0e38f, s = 0.f;
  for (int j = tt * 4; j < N_NODES; j += 64) {
    const int4   av = *(const int4*)  &adj[adjrow + j];
    const float4 ed = *(const float4*)&e_dst[j];
    const int   ai[4] = {av.x, av.y, av.z, av.w};
    const float ef[4] = {ed.x, ed.y, ed.z, ed.w};
#pragma unroll
    for (int i = 0; i < 4; ++i) {
      float v = es + ef[i];
      v = (v >= 0.f) ? v : ALPHA * v;
      v = (ai[i] > 0) ? v : NEG_INF;
      const float nm = fmaxf(m, v);
      s = s * __expf(m - nm) + __expf(v - nm);
      m = nm;
    }
  }
#pragma unroll
  for (int off = 8; off > 0; off >>= 1) {
    const float om = __shfl_xor(m, off, 16);
    const float os = __shfl_xor(s, off, 16);
    const float nm = fmaxf(m, om);
    s = s * __expf(m - nm) + os * __expf(om - nm);
    m = nm;
  }
  if (tt == 0) { sm_m[rr] = m; sm_s[rr] = s; }
  __syncthreads();

  // ---------------- Pass 2: f16 WMMA with register-prefetched tiles ----------
  const int wave = tid >> 5;
  const int lane = tid & 31;
  const int l16  = lane & 15;
  const bool hi  = lane >= 16;
  const int n0   = wave * 32;
  const float mrow = sm_m[rr];

  // B-fragment base pointers (16-bit B layout: V0..3 = K 0..7 / 8..15, V4..7 = +16)
  const f16* bcol0 = h16T + (size_t)(n0 + l16) * N_NODES + (hi ? 8 : 0);
  const f16* bcol1 = bcol0 + (size_t)16 * N_NODES;
  const f16* prow  = &sm_p16[l16][0];

  v8f acc0 = {};
  v8f acc1 = {};

  // prefetch registers: thread covers cols tt*4 + 64*q (q=0,1) of each BK tile
  int4   pav[2];
  float4 ped[2];
#pragma unroll
  for (int q = 0; q < 2; ++q) {
    const int c = tt * 4 + 64 * q;
    pav[q] = *(const int4*)  &adj[adjrow + c];
    ped[q] = *(const float4*)&e_dst[c];
  }

  for (int jb = 0; jb < N_NODES; jb += BK) {
    __syncthreads();   // previous tile fully consumed
    // build f16 probability tile from prefetched registers
#pragma unroll
    for (int q = 0; q < 2; ++q) {
      const int c = tt * 4 + 64 * q;
      const int   ai[4] = {pav[q].x, pav[q].y, pav[q].z, pav[q].w};
      const float ef[4] = {ped[q].x, ped[q].y, ped[q].z, ped[q].w};
#pragma unroll
      for (int i = 0; i < 4; ++i) {
        float v = es + ef[i];
        v = (v >= 0.f) ? v : ALPHA * v;
        const float p = (ai[i] > 0) ? __expf(v - mrow) : 0.f;
        sm_p16[rr][c + i] = (f16)p;
      }
    }
    // prefetch next tile while WMMA below runs
    if (jb + BK < N_NODES) {
#pragma unroll
      for (int q = 0; q < 2; ++q) {
        const int c = jb + BK + tt * 4 + 64 * q;
        pav[q] = *(const int4*)  &adj[adjrow + c];
        ped[q] = *(const float4*)&e_dst[c];
      }
    }
    __syncthreads();

#pragma unroll
    for (int k0 = 0; k0 < BK; k0 += 32) {
      // A fragment (16-bit A layout: V0..3 = K 0..7 / 8..15, V4..7 = +16)
      const v8h alo = *(const v8h*)(prow + k0 + (hi ? 8 : 0));
      const v8h ahi = *(const v8h*)(prow + k0 + 16 + (hi ? 8 : 0));
      const v16h a = __builtin_shufflevector(alo, ahi, 0, 1, 2, 3, 4, 5, 6, 7,
                                             8, 9, 10, 11, 12, 13, 14, 15);
      // B fragments: 8 f16 at k0+8h and 8 at k0+16+8h per lane from transposed h
      const v8h b0lo = *(const v8h*)(bcol0 + jb + k0);
      const v8h b0hi = *(const v8h*)(bcol0 + jb + k0 + 16);
      const v16h b0 = __builtin_shufflevector(b0lo, b0hi, 0, 1, 2, 3, 4, 5, 6, 7,
                                              8, 9, 10, 11, 12, 13, 14, 15);
      const v8h b1lo = *(const v8h*)(bcol1 + jb + k0);
      const v8h b1hi = *(const v8h*)(bcol1 + jb + k0 + 16);
      const v16h b1 = __builtin_shufflevector(b1lo, b1hi, 0, 1, 2, 3, 4, 5, 6, 7,
                                              8, 9, 10, 11, 12, 13, 14, 15);

      acc0 = __builtin_amdgcn_wmma_f32_16x16x32_f16(false, a, false, b0,
                                                    (short)0, acc0, false, false);
      acc1 = __builtin_amdgcn_wmma_f32_16x16x32_f16(false, a, false, b1,
                                                    (short)0, acc1, false, false);
    }
  }

  // fold 1/sumexp (f32) and store
  const int lr  = hi ? 8 : 0;
  const int col = n0 + l16;
#pragma unroll
  for (int r = 0; r < 8; ++r) {
    const float inv = 1.0f / sm_s[lr + r];
    out[(size_t)(r0 + lr + r) * F_OUT + col]      = acc0[r] * inv;
    out[(size_t)(r0 + lr + r) * F_OUT + col + 16] = acc1[r] * inv;
  }
}

// ---------------------------------------------------------------------------
extern "C" void kernel_launch(void* const* d_in, const int* in_sizes, int n_in,
                              void* d_out, int out_size, void* d_ws, size_t ws_size,
                              hipStream_t stream) {
  const float* x   = (const float*)d_in[0];   // (8192, 512)
  const int*   adj = (const int*)  d_in[1];   // (8192, 8192)
  const float* W   = (const float*)d_in[2];   // (512, 256)
  const float* a   = (const float*)d_in[3];   // (512, 1)
  float* out = (float*)d_out;                 // (8192, 256)

  char* ws = (char*)d_ws;
  float* h     = (float*)ws;                                        // 8 MB f32
  f16*   h16T  = (f16*)(ws + (size_t)N_NODES * F_OUT * 4);          // 4 MB f16 (transposed)
  float* e_src = (float*)(ws + (size_t)N_NODES * F_OUT * 6);        // 32 KB
  float* e_dst = e_src + N_NODES;                                   // 32 KB

  gat_xw  <<<N_NODES / 16, 256, 0, stream>>>(x, W, h, h16T);
  gat_ev  <<<N_NODES / 8,  256, 0, stream>>>(h, a, e_src, e_dst);
  gat_attn<<<N_NODES / 16, 256, 0, stream>>>(h16T, adj, e_src, e_dst, out);
}
